// RegisterAttention_20478404067335
// MI455X (gfx1250) — matrix operand order repaired and checked by reading, not compiled, not run on hardware
//
#include <hip/hip_runtime.h>
#include <hip/hip_bf16.h>

typedef _Float16 v16h __attribute__((ext_vector_type(16)));
typedef _Float16 v8h  __attribute__((ext_vector_type(8)));
typedef float    v8f  __attribute__((ext_vector_type(8)));

#define DIM   512
#define ATT   128
#define EPSF  1e-6f

__device__ __forceinline__ float wredsum(float v) {
#pragma unroll
  for (int m = 16; m >= 1; m >>= 1) v += __shfl_xor(v, m, 32);
  return v;
}
__device__ __forceinline__ unsigned encF(float f) {
  unsigned u = __float_as_uint(f);
  return (u & 0x80000000u) ? ~u : (u | 0x80000000u);
}
__device__ __forceinline__ float decF(unsigned u) {
  return (u & 0x80000000u) ? __uint_as_float(u & 0x7FFFFFFFu) : __uint_as_float(~u);
}
__device__ __forceinline__ float siluf(float x) { return x / (1.0f + __expf(-x)); }

// ---------------------------------------------------------------- init scratch
__global__ __launch_bounds__(256) void k_init(float* g, float* gsum, unsigned* gmaxU) {
  int t = threadIdx.x;
  for (int i = t; i < 2048; i += 256) g[i] = 0.f;
  if (t < 4) { gsum[t] = 0.f; gmaxU[t] = 0x007FFFFFu; /* enc(-inf) */ }
}

// ------------------------------------------- wq_eff[r*512+d] = Wq[:,r4:]·rk /2
__global__ __launch_bounds__(256) void k_prep_wq(const float* __restrict__ Wq,
                                                 const float* __restrict__ bq,
                                                 const float* __restrict__ rk,
                                                 float* __restrict__ wqe,
                                                 float* __restrict__ bqe) {
  int gid = blockIdx.x * 256 + threadIdx.x;   // 2048 total
  int r = gid >> 9, d = gid & 511;
  float acc = 0.f;
#pragma unroll
  for (int dd = 0; dd < 4; ++dd) acc += Wq[d * 16 + r * 4 + dd] * rk[r * 4 + dd];
  wqe[gid] = acc * 0.5f;
  if (gid < 4) {
    float ab = 0.f;
#pragma unroll
    for (int dd = 0; dd < 4; ++dd) ab += bq[gid * 4 + dd] * rk[gid * 4 + dd];
    bqe[gid] = ab * 0.5f;
  }
}

// ------------------------------- pass 1: logits[n][4] + global max (ordered u32)
__global__ __launch_bounds__(256) void k_logits_max(const float* __restrict__ h,
                                                    const float* __restrict__ wqe,
                                                    const float* __restrict__ bqe,
                                                    float* __restrict__ logits,
                                                    unsigned* __restrict__ gmaxU, int N) {
  __shared__ unsigned bmax[4];
  int tid = threadIdx.x, w = tid >> 5, lane = tid & 31, d0 = lane * 16;
  if (tid < 4) bmax[tid] = 0x007FFFFFu;
  __syncthreads();
  float4 bq4 = *(const float4*)bqe;
  float bqa[4] = {bq4.x, bq4.y, bq4.z, bq4.w};
  float wmax[4] = {-3.4e38f, -3.4e38f, -3.4e38f, -3.4e38f};
  for (int i = 0; i < 16; ++i) {
    int row = blockIdx.x * 128 + w * 16 + i;
    if (row >= N) break;
    const float4* hp = (const float4*)(h + (size_t)row * DIM + d0);
    float hr[16];
#pragma unroll
    for (int q = 0; q < 4; ++q) {
      float4 x = hp[q];
      hr[q * 4] = x.x; hr[q * 4 + 1] = x.y; hr[q * 4 + 2] = x.z; hr[q * 4 + 3] = x.w;
    }
    float lg[4];
#pragma unroll
    for (int r = 0; r < 4; ++r) {
      const float4* wp = (const float4*)(wqe + r * DIM + d0);
      float p = 0.f;
#pragma unroll
      for (int q = 0; q < 4; ++q) {
        float4 x = wp[q];
        p += hr[q * 4] * x.x + hr[q * 4 + 1] * x.y + hr[q * 4 + 2] * x.z + hr[q * 4 + 3] * x.w;
      }
      lg[r] = wredsum(p) + bqa[r];
      wmax[r] = fmaxf(wmax[r], lg[r]);
    }
    if (lane == 0) {
      float4 o = make_float4(lg[0], lg[1], lg[2], lg[3]);
      *(float4*)(logits + (size_t)row * 4) = o;
    }
  }
  if (lane == 0) {
#pragma unroll
    for (int r = 0; r < 4; ++r) atomicMax(&bmax[r], encF(wmax[r]));
  }
  __syncthreads();
  if (tid < 4) atomicMax(&gmaxU[tid], bmax[tid]);
}

// -------------------- pass 2: g[r][512] = sum_n exp(l-max)*h[n,:], gsum[r]=sum exp
__global__ __launch_bounds__(256) void k_accum(const float* __restrict__ h,
                                               const float* __restrict__ logits,
                                               const unsigned* __restrict__ gmaxU,
                                               float* __restrict__ g,
                                               float* __restrict__ gsum, int N) {
  __shared__ float gls[2048];
  __shared__ float wls[4];
  int tid = threadIdx.x, w = tid >> 5, lane = tid & 31, d0 = lane * 16;
  for (int i = tid; i < 2048; i += 256) gls[i] = 0.f;
  if (tid < 4) wls[tid] = 0.f;
  __syncthreads();
  float gm[4];
#pragma unroll
  for (int r = 0; r < 4; ++r) gm[r] = decF(gmaxU[r]);
  float acc[4][16];
#pragma unroll
  for (int r = 0; r < 4; ++r)
#pragma unroll
    for (int k = 0; k < 16; ++k) acc[r][k] = 0.f;
  float wacc[4] = {0.f, 0.f, 0.f, 0.f};
  for (int i = 0; i < 16; ++i) {
    int row = blockIdx.x * 128 + w * 16 + i;
    if (row >= N) break;
    float4 l4 = *(const float4*)(logits + (size_t)row * 4);
    float wv[4] = {__expf(l4.x - gm[0]), __expf(l4.y - gm[1]),
                   __expf(l4.z - gm[2]), __expf(l4.w - gm[3])};
    const float4* hp = (const float4*)(h + (size_t)row * DIM + d0);
    float hr[16];
#pragma unroll
    for (int q = 0; q < 4; ++q) {
      float4 x = hp[q];
      hr[q * 4] = x.x; hr[q * 4 + 1] = x.y; hr[q * 4 + 2] = x.z; hr[q * 4 + 3] = x.w;
    }
#pragma unroll
    for (int r = 0; r < 4; ++r) {
      wacc[r] += wv[r];
#pragma unroll
      for (int k = 0; k < 16; ++k) acc[r][k] += wv[r] * hr[k];
    }
  }
#pragma unroll
  for (int r = 0; r < 4; ++r)
#pragma unroll
    for (int k = 0; k < 16; ++k) atomicAdd(&gls[r * DIM + d0 + k], acc[r][k]);
  if (lane == 0)
#pragma unroll
    for (int r = 0; r < 4; ++r) atomicAdd(&wls[r], wacc[r]);
  __syncthreads();
  for (int i = tid; i < 2048; i += 256) atomicAdd(&g[i], gls[i]);
  if (tid < 4) atomicAdd(&gsum[tid], wls[tid]);
}

// --------------------------------------------- tiny MLP (single block): -> q,v
__device__ __forceinline__ void ln16(float* x, const float* __restrict__ s,
                                     const float* __restrict__ b) {
  float m = 0.f;
#pragma unroll
  for (int i = 0; i < 16; ++i) m += x[i];
  m *= (1.f / 16.f);
  float v = 0.f;
#pragma unroll
  for (int i = 0; i < 16; ++i) { float d = x[i] - m; v += d * d; }
  v *= (1.f / 16.f);
  float rs = rsqrtf(v + EPSF);
#pragma unroll
  for (int i = 0; i < 16; ++i) x[i] = (x[i] - m) * rs * s[i] + b[i];
}

__global__ __launch_bounds__(256) void k_mlp(const float* __restrict__ g,
                                             const float* __restrict__ gsum,
                                             const float* __restrict__ Wv, const float* __restrict__ bv,
                                             const float* __restrict__ lnregs, const float* __restrict__ lnregb,
                                             const float* __restrict__ ln1s, const float* __restrict__ ln1b,
                                             const float* __restrict__ W1, const float* __restrict__ b1,
                                             const float* __restrict__ W2, const float* __restrict__ b2,
                                             const float* __restrict__ W3, const float* __restrict__ b3,
                                             float* __restrict__ queries, float* __restrict__ valuesFlat) {
  __shared__ float reg16[16];
  __shared__ float r1[DIM];
  __shared__ float r2[DIM];
  int tid = threadIdx.x;
  if (tid < 16) {
    int r = tid >> 2;
    float acc = 0.f;
    for (int d = 0; d < DIM; ++d) acc += g[r * DIM + d] * Wv[d * 16 + tid];
    reg16[tid] = acc / gsum[r] + bv[tid];
  }
  __syncthreads();
  if (tid == 0) {
    float x[16];
#pragma unroll
    for (int i = 0; i < 16; ++i) x[i] = reg16[i];
    ln16(x, lnregs, lnregb);
    ln16(x, ln1s, ln1b);
#pragma unroll
    for (int i = 0; i < 16; ++i) reg16[i] = x[i];
  }
  __syncthreads();
  for (int j = tid; j < DIM; j += 256) {
    float acc = b1[j];
#pragma unroll
    for (int i = 0; i < 16; ++i) acc += reg16[i] * W1[i * DIM + j];
    r1[j] = siluf(acc);
  }
  __syncthreads();
  for (int j = tid; j < DIM; j += 256) {
    float acc = b2[j];
    for (int i = 0; i < DIM; ++i) acc += r1[i] * W2[i * DIM + j];
    r2[j] = siluf(acc);
  }
  __syncthreads();
  for (int j = tid; j < 1024; j += 256) {
    float acc = b3[j];
    for (int i = 0; i < DIM; ++i) acc += r2[i] * W3[i * 1024 + j];
    if (j < 512) queries[j] = acc; else valuesFlat[j - 512] = acc;
  }
}

// -------------------- wk_eff[r*512+d] = Wk[d, r*128:]·queries[r]/sqrt(128)
__global__ __launch_bounds__(256) void k_prep_wk(const float* __restrict__ Wk,
                                                 const float* __restrict__ bk,
                                                 const float* __restrict__ queries,
                                                 float* __restrict__ wke,
                                                 float* __restrict__ bkq) {
  const float sc = 0.088388347648318447f;  // 1/sqrt(128)
  int gid = blockIdx.x * 256 + threadIdx.x;  // 2048 total
  int r = gid >> 9, d = gid & 511;
  float acc = 0.f;
  const float* wrow = Wk + (size_t)d * DIM + r * ATT;
  const float* q = queries + r * ATT;
  for (int j = 0; j < ATT; ++j) acc += wrow[j] * q[j];
  wke[gid] = acc * sc;
  if (gid < 4) {
    float ab = 0.f;
    const float* bb = bk + gid * ATT;
    const float* qq = queries + gid * ATT;
    for (int j = 0; j < ATT; ++j) ab += bb[j] * qq[j];
    bkq[gid] = ab * sc;
  }
}

// -------- WmT[n*512+k] = (f16)Wm[k*512+n]  (K contiguous per column for B-frags)
__global__ __launch_bounds__(256) void k_prep_wmT(const float* __restrict__ Wm,
                                                  _Float16* __restrict__ WmT) {
  int gid = blockIdx.x * 256 + threadIdx.x;  // 262144 total
  int n = gid >> 9, k = gid & 511;
  WmT[gid] = (_Float16)Wm[(size_t)k * DIM + n];
}

// ====================== fused main kernel: 32 rows/block =====================
__global__ __launch_bounds__(256) void k_fused(const float* __restrict__ h,
                                               const float* __restrict__ wke,
                                               const float* __restrict__ bkq,
                                               const float* __restrict__ valuesFlat,
                                               const float* __restrict__ ln2s, const float* __restrict__ ln2b,
                                               const _Float16* __restrict__ WmT, const float* __restrict__ bm,
                                               const float* __restrict__ ln3s, const float* __restrict__ ln3b,
                                               float* __restrict__ out, int N) {
  extern __shared__ char smem[];
  float*    h2L  = (float*)smem;                                 // 32*512 f32  (64KB)
  _Float16* aL   = (_Float16*)(smem + 32 * DIM * 4);             // 32*512 f16  (32KB)
  float*    mlpL = (float*)(smem + 32 * DIM * 4 + 32 * DIM * 2); // 32*512 f32  (64KB)
  const int tid = threadIdx.x, w = tid >> 5, lane = tid & 31;
  const int d0 = lane * 16;
  const int rh = d0 >> 7;  // head index owned by this lane's 16-wide slice
  const long rowBase = (long)blockIdx.x * 32;

  float vals[16], s2v[16], b2v[16];
  {
    const float4* vp = (const float4*)(valuesFlat + d0);
    const float4* sp = (const float4*)(ln2s + d0);
    const float4* bp = (const float4*)(ln2b + d0);
#pragma unroll
    for (int q = 0; q < 4; ++q) {
      float4 a = vp[q], s = sp[q], b = bp[q];
      vals[q * 4] = a.x; vals[q * 4 + 1] = a.y; vals[q * 4 + 2] = a.z; vals[q * 4 + 3] = a.w;
      s2v[q * 4] = s.x;  s2v[q * 4 + 1] = s.y;  s2v[q * 4 + 2] = s.z;  s2v[q * 4 + 3] = s.w;
      b2v[q * 4] = b.x;  b2v[q * 4 + 1] = b.y;  b2v[q * 4 + 2] = b.z;  b2v[q * 4 + 3] = b.w;
    }
  }
  float4 bk4 = *(const float4*)bkq;
  float bka[4] = {bk4.x, bk4.y, bk4.z, bk4.w};

  // ---- stage 1: per-row head attention + residual + LN2 -> LDS (f32 + f16) --
#pragma unroll
  for (int i = 0; i < 4; ++i) {
    const int m = w * 4 + i;
    const long row = rowBase + m;
    float hr[16];
    if (row < N) {
      const float4* hp = (const float4*)(h + (size_t)row * DIM + d0);
#pragma unroll
      for (int q = 0; q < 4; ++q) {
        float4 x = hp[q];
        hr[q * 4] = x.x; hr[q * 4 + 1] = x.y; hr[q * 4 + 2] = x.z; hr[q * 4 + 3] = x.w;
      }
    } else {
#pragma unroll
      for (int k = 0; k < 16; ++k) hr[k] = 0.f;
    }
    float lg[4];
#pragma unroll
    for (int r = 0; r < 4; ++r) {
      const float4* wp = (const float4*)(wke + r * DIM + d0);
      float p = 0.f;
#pragma unroll
      for (int q = 0; q < 4; ++q) {
        float4 x = wp[q];
        p += hr[q * 4] * x.x + hr[q * 4 + 1] * x.y + hr[q * 4 + 2] * x.z + hr[q * 4 + 3] * x.w;
      }
      lg[r] = wredsum(p) + bka[r];
    }
    float mx = fmaxf(fmaxf(lg[0], lg[1]), fmaxf(lg[2], lg[3]));
    float e0 = __expf(lg[0] - mx), e1 = __expf(lg[1] - mx);
    float e2 = __expf(lg[2] - mx), e3 = __expf(lg[3] - mx);
    float inv = 1.f / (e0 + e1 + e2 + e3);
    float av[4] = {e0 * inv, e1 * inv, e2 * inv, e3 * inv};
    float a = av[rh];
    float t[16];
    float s1 = 0.f, sq = 0.f;
#pragma unroll
    for (int k = 0; k < 16; ++k) {
      t[k] = hr[k] + a * vals[k];
      s1 += t[k]; sq += t[k] * t[k];
    }
    s1 = wredsum(s1); sq = wredsum(sq);
    float mean = s1 * (1.f / 512.f);
    float var = sq * (1.f / 512.f) - mean * mean;
    float rstd = rsqrtf(var + EPSF);
#pragma unroll
    for (int k = 0; k < 16; ++k) {
      float h2 = (t[k] - mean) * rstd * s2v[k] + b2v[k];
      h2L[m * DIM + d0 + k] = h2;
      aL[m * DIM + d0 + k] = (_Float16)h2;
    }
  }
  __syncthreads();

  // ---- stage 2: mlp = silu(h2 @ Wm + bm) via v_wmma_f32_16x16x32_f16 --------
  // wave w owns 64 output cols [w*64, w*64+64); 2 row-tiles x 4 col-tiles.
  v8f acc[2][4];
#pragma unroll
  for (int rt = 0; rt < 2; ++rt)
#pragma unroll
    for (int ct = 0; ct < 4; ++ct)
#pragma unroll
      for (int e = 0; e < 8; ++e) acc[rt][ct][e] = 0.f;
  const int nbase = w * 64;
  const int mrow = lane & 15;
  const bool hiHalf = (lane >= 16);
#pragma unroll 4
  for (int kk = 0; kk < 16; ++kk) {
    const int kb = kk * 32;
    v16h Afrag[2];
#pragma unroll
    for (int rt = 0; rt < 2; ++rt) {
      // A frag (contiguous k, same order as B): lanes0-15 (M=mrow) K=[0..15]; lanes16-31 K=[16..31]
      const _Float16* ap = aL + (rt * 16 + mrow) * DIM + kb + (hiHalf ? 16 : 0);
      v8h lo = *(const v8h*)ap;
      v8h hi = *(const v8h*)(ap + 8);
      Afrag[rt] = __builtin_shufflevector(lo, hi, 0, 1, 2, 3, 4, 5, 6, 7,
                                          8, 9, 10, 11, 12, 13, 14, 15);
    }
#pragma unroll
    for (int ct = 0; ct < 4; ++ct) {
      // B layout: lanes0-15 (N=mrow) hold K=kb..kb+15; lanes16-31 K=kb+16..kb+31
      const int n = nbase + ct * 16 + mrow;
      const int ks = kb + (hiHalf ? 16 : 0);
      v16h B = *(const v16h*)(WmT + (size_t)n * DIM + ks);
      acc[0][ct] = __builtin_amdgcn_wmma_f32_16x16x32_f16(false, Afrag[0], false, B,
                                                          (short)0, acc[0][ct], false, false);
      acc[1][ct] = __builtin_amdgcn_wmma_f32_16x16x32_f16(false, Afrag[1], false, B,
                                                          (short)0, acc[1][ct], false, false);
    }
  }
  // epilogue: +bm, silu, scatter C frags (N=lane%16, M=vgpr + 8*(lane>=16)) to LDS
  const int mofs = hiHalf ? 8 : 0;
#pragma unroll
  for (int ct = 0; ct < 4; ++ct) {
    const int n = nbase + ct * 16 + mrow;
    const float bmn = bm[n];
#pragma unroll
    for (int rt = 0; rt < 2; ++rt) {
#pragma unroll
      for (int v = 0; v < 8; ++v) {
        float x = siluf(acc[rt][ct][v] + bmn);
        int m = rt * 16 + v + mofs;
        mlpL[m * DIM + n] = x;
      }
    }
  }
  __syncthreads();

  // ---- stage 3: h3 = LN3(h2 + mlp) -> global ------------------------------
  float s3v[16], b3v[16];
  {
    const float4* sp = (const float4*)(ln3s + d0);
    const float4* bp = (const float4*)(ln3b + d0);
#pragma unroll
    for (int q = 0; q < 4; ++q) {
      float4 s = sp[q], b = bp[q];
      s3v[q * 4] = s.x; s3v[q * 4 + 1] = s.y; s3v[q * 4 + 2] = s.z; s3v[q * 4 + 3] = s.w;
      b3v[q * 4] = b.x; b3v[q * 4 + 1] = b.y; b3v[q * 4 + 2] = b.z; b3v[q * 4 + 3] = b.w;
    }
  }
#pragma unroll
  for (int i = 0; i < 4; ++i) {
    const int m = w * 4 + i;
    const long row = rowBase + m;
    float t[16];
    float s1 = 0.f, sq = 0.f;
    const float4* xp = (const float4*)(h2L + m * DIM + d0);
    const float4* yp = (const float4*)(mlpL + m * DIM + d0);
#pragma unroll
    for (int q = 0; q < 4; ++q) {
      float4 x = xp[q], y = yp[q];
      t[q * 4] = x.x + y.x; t[q * 4 + 1] = x.y + y.y;
      t[q * 4 + 2] = x.z + y.z; t[q * 4 + 3] = x.w + y.w;
    }
#pragma unroll
    for (int k = 0; k < 16; ++k) { s1 += t[k]; sq += t[k] * t[k]; }
    s1 = wredsum(s1); sq = wredsum(sq);
    float mean = s1 * (1.f / 512.f);
    float var = sq * (1.f / 512.f) - mean * mean;
    float rstd = rsqrtf(var + EPSF);
    if (row < N) {
      float* op = out + (size_t)row * DIM + d0;
#pragma unroll
      for (int q = 0; q < 4; ++q) {
        float4 o;
        o.x = (t[q * 4]     - mean) * rstd * s3v[q * 4]     + b3v[q * 4];
        o.y = (t[q * 4 + 1] - mean) * rstd * s3v[q * 4 + 1] + b3v[q * 4 + 1];
        o.z = (t[q * 4 + 2] - mean) * rstd * s3v[q * 4 + 2] + b3v[q * 4 + 2];
        o.w = (t[q * 4 + 3] - mean) * rstd * s3v[q * 4 + 3] + b3v[q * 4 + 3];
        *(float4*)(op + q * 4) = o;
      }
    }
  }
}

// ============================================================================
extern "C" void kernel_launch(void* const* d_in, const int* in_sizes, int n_in,
                              void* d_out, int out_size, void* d_ws, size_t ws_size,
                              hipStream_t stream) {
  const float* h      = (const float*)d_in[0];
  // d_in[1] = spin (unused by the reference)
  const float* rk     = (const float*)d_in[2];
  const float* Wq     = (const float*)d_in[3];
  const float* bq     = (const float*)d_in[4];
  const float* Wv     = (const float*)d_in[5];
  const float* bv     = (const float*)d_in[6];
  const float* lnregs = (const float*)d_in[7];
  const float* lnregb = (const float*)d_in[8];
  const float* ln1s   = (const float*)d_in[9];
  const float* ln1b   = (const float*)d_in[10];
  const float* W1     = (const float*)d_in[11];
  const float* b1     = (const float*)d_in[12];
  const float* W2     = (const float*)d_in[13];
  const float* b2     = (const float*)d_in[14];
  const float* W3     = (const float*)d_in[15];
  const float* b3     = (const float*)d_in[16];
  const float* Wk     = (const float*)d_in[17];
  const float* bk     = (const float*)d_in[18];
  const float* ln2s   = (const float*)d_in[19];
  const float* ln2b   = (const float*)d_in[20];
  const float* Wm     = (const float*)d_in[21];
  const float* bm     = (const float*)d_in[22];
  const float* ln3s   = (const float*)d_in[23];
  const float* ln3b   = (const float*)d_in[24];

  const int N = in_sizes[0] / DIM;  // 100000

  // workspace layout (float offsets)
  float* ws = (float*)d_ws;
  float*    wqe        = ws + 0;         // 2048
  float*    bqe        = ws + 2048;      // 4
  float*    bkq        = ws + 2052;      // 4
  float*    gsum       = ws + 2056;      // 4
  unsigned* gmaxU      = (unsigned*)(ws + 2060);  // 4
  float*    g          = ws + 2064;      // 2048
  float*    logits     = ws + 4112;      // N*4
  float*    queries    = ws + 4112 + 400000;           // 512
  float*    valuesFlat = ws + 4112 + 400000 + 512;     // 512
  float*    wke        = ws + 4112 + 400000 + 1024;    // 2048
  _Float16* WmT        = (_Float16*)(ws + 4112 + 400000 + 1024 + 2048);  // 262144 halves

  const int gridP = (N + 127) / 128;
  const int gridF = (N + 31) / 32;

  k_init<<<1, 256, 0, stream>>>(g, gsum, gmaxU);
  k_prep_wq<<<8, 256, 0, stream>>>(Wq, bq, rk, wqe, bqe);
  k_prep_wmT<<<1024, 256, 0, stream>>>(Wm, WmT);
  k_logits_max<<<gridP, 256, 0, stream>>>(h, wqe, bqe, logits, gmaxU, N);
  k_accum<<<gridP, 256, 0, stream>>>(h, logits, gmaxU, g, gsum, N);
  k_mlp<<<1, 256, 0, stream>>>(g, gsum, Wv, bv, lnregs, lnregb, ln1s, ln1b,
                               W1, b1, W2, b2, W3, b3, queries, valuesFlat);
  k_prep_wk<<<8, 256, 0, stream>>>(Wk, bk, queries, wke, bkq);

  const int ldsBytes = 32 * DIM * 4 + 32 * DIM * 2 + 32 * DIM * 4;  // 160 KB
  (void)hipFuncSetAttribute((const void*)k_fused,
                            hipFuncAttributeMaxDynamicSharedMemorySize, ldsBytes);
  k_fused<<<gridF, 256, ldsBytes, stream>>>(h, wke, bkq, valuesFlat, ln2s, ln2b,
                                            WmT, bm, ln3s, ln3b, (float*)d_out, N);
}
